// MultiHeadAttention_45732811768341
// MI455X (gfx1250) — matrix operand order repaired and checked by reading, not compiled, not run on hardware
//
#include <hip/hip_runtime.h>
#include <hip/hip_bf16.h>

typedef __attribute__((ext_vector_type(16))) _Float16 v16h;
typedef __attribute__((ext_vector_type(8)))  float    v8f;
typedef __attribute__((ext_vector_type(4)))  unsigned uv4;
typedef __attribute__((ext_vector_type(8)))  int      iv8;
typedef __attribute__((ext_vector_type(4)))  int      iv4;

#define HIDDEN   1024
#define HEADS    16
#define HEAD_DIM 64
#define BATCH    2
#define SEQ      2048
#define MTOT     (BATCH * SEQ)

union FragH { v16h h; unsigned u[8]; };

__device__ inline v8f wmma_f16(const v16h& a, const v16h& b, const v8f& c) {
  return __builtin_amdgcn_wmma_f32_16x16x32_f16(false, a, false, b, (short)0, c,
                                                false, false);
}

__device__ inline v8f zero8() {
  v8f z;
  for (int i = 0; i < 8; ++i) z[i] = 0.0f;
  return z;
}

// TDM 2-D tile load: global (row-major, f16) -> LDS, with LDS row padding.
//   tile_d0 = contiguous elements per row, tile_d1 = rows,
//   stride  = global row stride (elements),
//   pad_interval code c -> pad every 2^(c+1) DWORDs, pad_amount code a -> a+1 DWORDs.
__device__ inline void tdm_load_2d(unsigned lds_off, const void* gaddr,
                                   unsigned tile_d0, unsigned tile_d1,
                                   unsigned stride, unsigned pad_interval,
                                   unsigned pad_amount)
{
  size_t ga = (size_t)gaddr;
  uv4 g0;
  g0[0] = 1u;                                            // count=1, user mode
  g0[1] = lds_off;                                       // lds_addr (bytes)
  g0[2] = (unsigned)(ga & 0xFFFFFFFFu);                  // global_addr[31:0]
  g0[3] = (unsigned)((ga >> 32) & 0x01FFFFFFu) | (2u << 30); // addr[56:32], type=2
  iv8 g1;
  g1[0] = (int)((1u << 16) |                             // data_size = 2 bytes
                (1u << 20) |                             // pad_enable
                (pad_interval << 22) | (pad_amount << 25));
  g1[1] = (int)((tile_d0 & 0xFFFFu) << 16);              // tensor_dim0 = tile_d0
  g1[2] = (int)((tile_d1 & 0xFFFFu) << 16);              // tensor_dim1 = tile_d1
  g1[3] = (int)((tile_d0 & 0xFFFFu) << 16);              // tile_dim0
  g1[4] = (int)(tile_d1 & 0xFFFFu);                      // tile_dim1 (tile_dim2=0)
  g1[5] = (int)stride;                                   // tensor_dim0_stride lo
  g1[6] = 0;
  g1[7] = 0;
  iv4 z4 = {0, 0, 0, 0};                                 // 2-D: groups 2/3 unused
  iv8 z8 = {0, 0, 0, 0, 0, 0, 0, 0};
  // clang-23 / therock toolchain: 6-arg form (v4u, v8i, v4i, v4i, v8i, i32 cpol)
  __builtin_amdgcn_tensor_load_to_lds(g0, g1, z4, z4, z8, 0);
}

// ---------------------------------------------------------------------------
// Projection GEMM: out[m][n] = sum_k A[m][k] * W[k][n] + bias[n]
// M=4096, N=K=1024. Block tile 64x64, BK=32, 4 waves (2x2), each wave 32x32.
// MODE 0: f16 out at [(b*H+h)*S + q][d]      (Q / K layout, per-head row-major)
// MODE 1: f16 out at [(b*H+h)*D + d][q]      (V transposed per head)
// MODE 2: f32 out at [m][n]                  (final output projection)
// ---------------------------------------------------------------------------
template <typename AT, int MODE>
__global__ __launch_bounds__(128)
void proj_gemm(const AT* __restrict__ A, const float* __restrict__ W,
               const float* __restrict__ bias, void* __restrict__ outp)
{
  __shared__ __attribute__((aligned(16))) _Float16 As[64 * 40]; // [m][k], pad
  __shared__ __attribute__((aligned(16))) _Float16 Bs[64 * 40]; // [n][k], pad

  const int tid  = threadIdx.x;
  const int lane = tid & 31;
  const int wave = tid >> 5;
  const int g    = lane >> 4;   // half-wave group
  const int lr   = lane & 15;
  const int wm   = wave >> 1;   // wave row 0..1
  const int wn   = wave & 1;    // wave col 0..1
  const int n0   = blockIdx.x * 64;
  const int m0   = blockIdx.y * 64;

  v8f acc[2][2];
  for (int i = 0; i < 2; ++i)
    for (int j = 0; j < 2; ++j) acc[i][j] = zero8();

  for (int k0 = 0; k0 < HIDDEN; k0 += 32) {
    __syncthreads();
    // A tile: 64 rows x 32 k, convert to f16
    for (int i = 0; i < 16; ++i) {
      int idx = i * 128 + tid;
      int r = idx >> 5, c = idx & 31;
      As[r * 40 + c] = (_Float16)A[(size_t)(m0 + r) * HIDDEN + (k0 + c)];
    }
    // B tile transposed: Bs[n][k] = W[(k0+k)*N + n0+n]
    for (int i = 0; i < 16; ++i) {
      int idx = i * 128 + tid;
      int kk = idx >> 6, n = idx & 63;
      Bs[n * 40 + kk] = (_Float16)W[(size_t)(k0 + kk) * HIDDEN + (n0 + n)];
    }
    __syncthreads();

    FragH af[2], bf[2];
    for (int mt = 0; mt < 2; ++mt) {
      int row = wm * 32 + mt * 16 + lr;
      for (int v = 0; v < 8; ++v) {
        int kA = (v < 4) ? (8 * g + 2 * v) : (16 + 8 * g + 2 * (v - 4));
        af[mt].u[v] = *(const unsigned*)&As[row * 40 + kA];
      }
    }
    for (int nt = 0; nt < 2; ++nt) {
      int row = wn * 32 + nt * 16 + lr;
      for (int v = 0; v < 8; ++v) {
        int kB = (v < 4) ? (8 * g + 2 * v) : (16 + 8 * g + 2 * (v - 4));
        bf[nt].u[v] = *(const unsigned*)&Bs[row * 40 + kB];
      }
    }
    for (int mt = 0; mt < 2; ++mt)
      for (int nt = 0; nt < 2; ++nt)
        acc[mt][nt] = wmma_f16(af[mt].h, bf[nt].h, acc[mt][nt]);
  }

  // Epilogue: C layout = lane lr -> N, VGPR r -> M = r + 8*g
  for (int mt = 0; mt < 2; ++mt) {
    for (int nt = 0; nt < 2; ++nt) {
      for (int r = 0; r < 8; ++r) {
        int m = m0 + wm * 32 + mt * 16 + r + 8 * g;
        int n = n0 + wn * 32 + nt * 16 + lr;
        float val = acc[mt][nt][r] + bias[n];
        if (MODE == 2) {
          ((float*)outp)[(size_t)m * HIDDEN + n] = val;
        } else {
          int b = m >> 11, q = m & (SEQ - 1);
          int h = n >> 6,  d = n & (HEAD_DIM - 1);
          size_t addr = (MODE == 0)
            ? ((size_t)((b * HEADS + h) * SEQ + q) * HEAD_DIM + d)
            : ((size_t)((b * HEADS + h) * HEAD_DIM + d) * SEQ + q);
          ((_Float16*)outp)[addr] = (_Float16)val;
        }
      }
    }
  }
}

// ---------------------------------------------------------------------------
// Flash-style attention, TDM double-buffered. Block = 4 waves = 64 queries
// for one (b,h). Wave 0 issues tensor_load_to_lds for the next 32-key tile
// while all waves compute the current one; gate with s_wait_tensorcnt.
// Q/K stored [bh][s][dh] f16; V stored [bh][dh][s] f16; ctx out [B*S][1024] f16.
// ---------------------------------------------------------------------------
__global__ __launch_bounds__(128)
void attn_kernel(const _Float16* __restrict__ Q, const _Float16* __restrict__ K,
                 const _Float16* __restrict__ V, _Float16* __restrict__ ctx)
{
  __shared__ __attribute__((aligned(16))) _Float16 Kt[2][32 * 72]; // [key][dh]
  __shared__ __attribute__((aligned(16))) _Float16 Vt[2][64 * 40]; // [dh][key]
  __shared__ __attribute__((aligned(16))) _Float16 Pt[4][16 * 40]; // per-wave P

  const int tid   = threadIdx.x;
  const int lane  = tid & 31;
  const int wave  = tid >> 5;
  const int g     = lane >> 4;
  const int lr    = lane & 15;
  const int bh    = blockIdx.y;
  const int qbase = blockIdx.x * 64 + wave * 16;

  const _Float16* Qp = Q + (size_t)bh * SEQ * HEAD_DIM;
  const _Float16* Kp = K + (size_t)bh * SEQ * HEAD_DIM;
  const _Float16* Vp = V + (size_t)bh * HEAD_DIM * SEQ;

  const unsigned ktOff[2] = { (unsigned)(size_t)&Kt[0][0],
                              (unsigned)(size_t)&Kt[1][0] };
  const unsigned vtOff[2] = { (unsigned)(size_t)&Vt[0][0],
                              (unsigned)(size_t)&Vt[1][0] };

  // Load the wave's 16x64 Q tile as two 16x32 A fragments (held all loop long)
  FragH qa[2];
  for (int c = 0; c < 2; ++c)
    for (int v = 0; v < 8; ++v) {
      int k = c * 32 + ((v < 4) ? (8 * g + 2 * v) : (16 + 8 * g + 2 * (v - 4)));
      qa[c].u[v] = *(const unsigned*)&Qp[(size_t)(qbase + lr) * HEAD_DIM + k];
    }

  v8f acc[4];
  for (int dn = 0; dn < 4; ++dn) acc[dn] = zero8();
  float mrow[8], lrow[8];
  for (int r = 0; r < 8; ++r) { mrow[r] = -INFINITY; lrow[r] = 0.0f; }

  const int T = SEQ / 32;

  // Prologue: TDM tile 0 into buffer 0.
  if (wave == 0) {
    // K tile: 32 rows x 64 halfs, row 128B (32 DW) -> pad code 4, +4 DW pad
    tdm_load_2d(ktOff[0], Kp, HEAD_DIM, 32, HEAD_DIM, 4, 3);
    // V tile: 64 rows x 32 halfs, row 64B (16 DW) -> pad code 3, +4 DW pad
    tdm_load_2d(vtOff[0], Vp, 32, HEAD_DIM, SEQ, 3, 3);
  }

  for (int kt = 0; kt < T; ++kt) {
    const int p = kt & 1;

    if (wave == 0) {
      if (kt + 1 < T) {
        const size_t kb = (size_t)(kt + 1) * 32;
        tdm_load_2d(ktOff[1 - p], Kp + kb * HEAD_DIM, HEAD_DIM, 32, HEAD_DIM, 4, 3);
        tdm_load_2d(vtOff[1 - p], Vp + kb, 32, HEAD_DIM, SEQ, 3, 3);
        __builtin_amdgcn_s_wait_tensorcnt(2);  // tile kt's 2 ops done (in-order)
      } else {
        __builtin_amdgcn_s_wait_tensorcnt(0);
      }
    }
    __syncthreads();   // buffer p ready for every wave

    // Scores: S = Q K^T for two 16-key N-tiles, dh reduced in two chunks
    v8f s[2];
    s[0] = zero8(); s[1] = zero8();
    for (int nt = 0; nt < 2; ++nt) {
      for (int c = 0; c < 2; ++c) {
        FragH bf;
        for (int v = 0; v < 8; ++v)
          bf.u[v] = *(const unsigned*)
              &Kt[p][(nt * 16 + lr) * 72 + c * 32 + ((v < 4) ? (8 * g + 2 * v) : (16 + 8 * g + 2 * (v - 4)))];
        s[nt] = wmma_f16(qa[c].h, bf.h, s[nt]);
      }
    }

    // Online softmax per row M = r + 8*g (stats replicated across 16 lanes)
    for (int r = 0; r < 8; ++r) {
      float s0 = s[0][r] * 0.125f;   // 1/sqrt(64)
      float s1 = s[1][r] * 0.125f;
      float t = fmaxf(s0, s1);
      t = fmaxf(t, __shfl_xor(t, 1));
      t = fmaxf(t, __shfl_xor(t, 2));
      t = fmaxf(t, __shfl_xor(t, 4));
      t = fmaxf(t, __shfl_xor(t, 8));
      float mn    = fmaxf(mrow[r], t);
      float alpha = __expf(mrow[r] - mn);
      float p0 = __expf(s0 - mn), p1 = __expf(s1 - mn);
      float rs = p0 + p1;
      rs += __shfl_xor(rs, 1);
      rs += __shfl_xor(rs, 2);
      rs += __shfl_xor(rs, 4);
      rs += __shfl_xor(rs, 8);
      lrow[r] = lrow[r] * alpha + rs;
      mrow[r] = mn;
      for (int dn = 0; dn < 4; ++dn) acc[dn][r] *= alpha;
      // C-layout -> LDS so P can be re-read in A-fragment layout
      Pt[wave][(r + 8 * g) * 40 + lr]      = (_Float16)p0;
      Pt[wave][(r + 8 * g) * 40 + 16 + lr] = (_Float16)p1;
    }

    FragH pf;
    for (int v = 0; v < 8; ++v) {
      int k = (v < 4) ? (8 * g + 2 * v) : (16 + 8 * g + 2 * (v - 4));
      pf.u[v] = *(const unsigned*)&Pt[wave][lr * 40 + k];
    }
    for (int dn = 0; dn < 4; ++dn) {
      FragH vf;
      for (int v = 0; v < 8; ++v)
        vf.u[v] = *(const unsigned*)&Vt[p][(dn * 16 + lr) * 40 + ((v < 4) ? (8 * g + 2 * v) : (16 + 8 * g + 2 * (v - 4)))];
      acc[dn] = wmma_f16(pf.h, vf.h, acc[dn]);
    }

    __syncthreads();   // all waves done reading buffer p (DScnt drained)
  }

  // Normalize and write ctx as f16 [B*S][HIDDEN] (feeds output projection)
  const int b = bh >> 4, h = bh & 15;
  for (int r = 0; r < 8; ++r) {
    float inv = 1.0f / lrow[r];
    int q = qbase + r + 8 * g;
    size_t rowtok = (size_t)(b * SEQ + q) * HIDDEN;
    for (int dn = 0; dn < 4; ++dn) {
      int col = h * HEAD_DIM + dn * 16 + lr;
      ctx[rowtok + col] = (_Float16)(acc[dn][r] * inv);
    }
  }
}

// ---------------------------------------------------------------------------
extern "C" void kernel_launch(void* const* d_in, const int* in_sizes, int n_in,
                              void* d_out, int out_size, void* d_ws, size_t ws_size,
                              hipStream_t stream)
{
  (void)in_sizes; (void)n_in; (void)out_size; (void)ws_size;
  const float* x  = (const float*)d_in[0];
  const float* Wq = (const float*)d_in[1];
  const float* bq = (const float*)d_in[2];
  const float* Wk = (const float*)d_in[3];
  const float* bk = (const float*)d_in[4];
  const float* Wv = (const float*)d_in[5];
  const float* bv = (const float*)d_in[6];
  const float* Wo = (const float*)d_in[7];
  const float* bo = (const float*)d_in[8];

  const size_t qkv_elems = (size_t)BATCH * HEADS * SEQ * HEAD_DIM; // 4.19M
  _Float16* q_ws = (_Float16*)d_ws;
  _Float16* k_ws = q_ws + qkv_elems;
  _Float16* v_ws = k_ws + qkv_elems;
  _Float16* c_ws = v_ws + qkv_elems;   // ctx [B*S][HIDDEN]; total 32 MB of ws

  dim3 gemm_grid(HIDDEN / 64, MTOT / 64);   // (16, 64)
  proj_gemm<float, 0><<<gemm_grid, 128, 0, stream>>>(x, Wq, bq, (void*)q_ws);
  proj_gemm<float, 0><<<gemm_grid, 128, 0, stream>>>(x, Wk, bk, (void*)k_ws);
  proj_gemm<float, 1><<<gemm_grid, 128, 0, stream>>>(x, Wv, bv, (void*)v_ws);

  attn_kernel<<<dim3(SEQ / 64, BATCH * HEADS), 128, 0, stream>>>(
      q_ws, k_ws, v_ws, c_ws);

  proj_gemm<_Float16, 2><<<gemm_grid, 128, 0, stream>>>(c_ws, Wo, bo, d_out);
}
